// MaskClipHead_83708912599559
// MI455X (gfx1250) — compile-verified, matrix operand order checked by reading, not run on hardware
//
#include <hip/hip_runtime.h>
#include <hip/hip_bf16.h>
#include <math.h>

#define BB 32
#define NN 4096
#define CC 768
#define TT 1000
#define KK 20
#define KP 32   // K padded to 32 for WMMA

typedef __attribute__((ext_vector_type(16))) __bf16 v16bf;
typedef __attribute__((ext_vector_type(8)))  __bf16 v8bf;
typedef __attribute__((ext_vector_type(8)))  float  v8f;

// Pointee type expected by __builtin_amdgcn_global_load_async_to_lds_b128
// (from the compiler diagnostic: 'int __attribute__((vector_size(16)))').
typedef int v4i __attribute__((vector_size(16)));
typedef __attribute__((address_space(1))) v4i as1_v4i;   // global
typedef __attribute__((address_space(3))) v4i as3_v4i;   // LDS

#if __has_builtin(__builtin_amdgcn_global_load_async_to_lds_b128)
#define HAVE_ASYNC_LDS 1
#else
#define HAVE_ASYNC_LDS 0
#endif

// Copy n16 * 16 bytes from global (16B-aligned) into LDS, cooperatively.
// Uses the CDNA5 async-to-LDS path (ASYNCcnt) when the toolchain exposes it.
__device__ __forceinline__ void lds_copy_b128(__bf16* dst, const __bf16* src,
                                              int n16, int tid, int nthreads) {
#if HAVE_ASYNC_LDS
    for (int e = tid; e < n16; e += nthreads) {
        __builtin_amdgcn_global_load_async_to_lds_b128(
            (as1_v4i*)(const void*)(src + e * 8),
            (as3_v4i*)(void*)(dst + e * 8), 0, 0);
    }
#if __has_builtin(__builtin_amdgcn_s_wait_asynccnt)
    __builtin_amdgcn_s_wait_asynccnt(0);
#else
    asm volatile("s_wait_asynccnt 0x0" ::: "memory");
#endif
#else
    const v8bf* s = reinterpret_cast<const v8bf*>(src);
    v8bf* d = reinterpret_cast<v8bf*>(dst);
    for (int e = tid; e < n16; e += nthreads) d[e] = s[e];
#endif
}

// Load a 16-element bf16 WMMA fragment (A-layout, row pointer pre-applied).
// ISA 7.12.2: lanes 0-15 hold k={0..7,16..23}, lanes 16-31 k={8..15,24..31}.
__device__ __forceinline__ v16bf load_frag16(const __bf16* p, int g) {
    v8bf lo = *reinterpret_cast<const v8bf*>(p + 8 * g);
    v8bf hi = *reinterpret_cast<const v8bf*>(p + 16 + 8 * g);
    v16bf r;
#pragma unroll
    for (int i = 0; i < 8; ++i) { r[i] = lo[i]; r[8 + i] = hi[i]; }
    return r;
}

// ---------------------------------------------------------------------------
// K1: vg_logit = g_feat @ text^T ; top-20 ; gather agg_text (both bf16
// layouts, zero-padded to 32 rows) ; zero denom.  One block per image.
// ---------------------------------------------------------------------------
__global__ __launch_bounds__(256) void k1_topk_gather(
    const float* __restrict__ g_feat,   // [B,1,C]
    const float* __restrict__ text,     // [T,C]
    __bf16* __restrict__ aggBF,         // [B,KP,C] bf16 (rows >=K zero)
    __bf16* __restrict__ aggBT,         // [B,C,KP] bf16 (cols >=K zero)
    float* __restrict__ denom)          // [B,K]
{
    __shared__ float gvec[CC];
    __shared__ float logit[TT];
    __shared__ float redv[256];
    __shared__ int   redi[256];
    __shared__ int   topk[KK];
    const int b = blockIdx.x, tid = threadIdx.x;

    for (int c = tid; c < CC; c += 256) gvec[c] = g_feat[b * CC + c];
    __syncthreads();

    for (int t = tid; t < TT; t += 256) {
        float s = 0.f;
        const float* tr = text + (size_t)t * CC;
        for (int c = 0; c < CC; ++c) s += gvec[c] * tr[c];
        logit[t] = s;
    }
    __syncthreads();

    for (int i = 0; i < KK; ++i) {
        float v = -INFINITY; int vi = TT;
        for (int t = tid; t < TT; t += 256) {
            float l = logit[t];
            if (l > v || (l == v && t < vi)) { v = l; vi = t; }
        }
        redv[tid] = v; redi[tid] = vi;
        __syncthreads();
        for (int s = 128; s > 0; s >>= 1) {
            if (tid < s) {
                float ov = redv[tid + s]; int oi = redi[tid + s];
                if (ov > redv[tid] || (ov == redv[tid] && oi < redi[tid])) {
                    redv[tid] = ov; redi[tid] = oi;
                }
            }
            __syncthreads();
        }
        if (tid == 0) { topk[i] = redi[0]; logit[redi[0]] = -INFINITY; }
        __syncthreads();
    }

    for (int e = tid; e < KP * CC; e += 256) {
        int k = e / CC, c = e - k * CC;
        float val = (k < KK) ? text[(size_t)topk[k] * CC + c] : 0.f;
        aggBF[(size_t)b * KP * CC + e] = (__bf16)val;
        aggBT[(size_t)b * CC * KP + (size_t)c * KP + k] = (__bf16)val;
    }
    if (tid < KK) denom[b * KK + tid] = 0.f;
}

// ---------------------------------------------------------------------------
// K2: seg[b,k,n] = exp(tau) * sum_c agg[b,k,c]*feat[b,n,c]   (bf16 WMMA)
// grid: (N/128, B); 8 waves, each owns 16 n-rows.  B-operand fragments are
// loaded straight from global (per-lane row-contiguous) -> no loop barriers.
// ---------------------------------------------------------------------------
__global__ __launch_bounds__(256) void k2_seg_logit(
    const float* __restrict__ feat,     // [B,N,C]
    const __bf16* __restrict__ aggBF,   // [B,KP,C] bf16
    const float* __restrict__ tau,      // [1]
    float* __restrict__ seg)            // [B,K,N]
{
    __shared__ __align__(16) __bf16 aggT[KP * CC];   // 48 KB A-tile
    const int b = blockIdx.y;
    const int n0 = blockIdx.x * 128;
    const int tid = threadIdx.x;
    const int lane = tid & 31, wave = tid >> 5;
    const int g = lane >> 4, lrow = lane & 15;

    lds_copy_b128(aggT, aggBF + (size_t)b * KP * CC, KP * CC / 8, tid, 256);
    const float s = __expf(tau[0]);
    __syncthreads();

    // this lane's feat row (the n-column it feeds into the B fragment)
    const float4* rp = reinterpret_cast<const float4*>(
        feat + ((size_t)b * NN + n0 + wave * 16 + lrow) * CC);

    v8f acc0 = {}; v8f acc1 = {};
    for (int k0 = 0; k0 < CC; k0 += 32) {
        const int q = (k0 >> 2) + 2 * g;
        float4 f0 = rp[q];
        float4 f1 = rp[q + 1];
        float4 f2 = rp[q + 4];
        float4 f3 = rp[q + 5];
        v16bf bfrag;
        bfrag[0]  = (__bf16)f0.x; bfrag[1]  = (__bf16)f0.y;
        bfrag[2]  = (__bf16)f0.z; bfrag[3]  = (__bf16)f0.w;
        bfrag[4]  = (__bf16)f1.x; bfrag[5]  = (__bf16)f1.y;
        bfrag[6]  = (__bf16)f1.z; bfrag[7]  = (__bf16)f1.w;
        bfrag[8]  = (__bf16)f2.x; bfrag[9]  = (__bf16)f2.y;
        bfrag[10] = (__bf16)f2.z; bfrag[11] = (__bf16)f2.w;
        bfrag[12] = (__bf16)f3.x; bfrag[13] = (__bf16)f3.y;
        bfrag[14] = (__bf16)f3.z; bfrag[15] = (__bf16)f3.w;

        v16bf a0 = load_frag16(&aggT[lrow * CC + k0], g);
        v16bf a1 = load_frag16(&aggT[(16 + lrow) * CC + k0], g);
        acc0 = __builtin_amdgcn_wmma_f32_16x16x32_bf16(false, a0, false, bfrag,
                                                       (short)0, acc0, false, false);
        acc1 = __builtin_amdgcn_wmma_f32_16x16x32_bf16(false, a1, false, bfrag,
                                                       (short)0, acc1, false, false);
    }

    const int ncol = n0 + wave * 16 + lrow;
#pragma unroll
    for (int v = 0; v < 8; ++v) {
        int m = g * 8 + v;                 // acc0 -> classes 0..15
        if (m < KK)
            seg[((size_t)b * KK + m) * NN + ncol] = acc0[v] * s;
        int m1 = 16 + m;                   // acc1 -> classes 16..31 (only <20 kept)
        if (m1 < KK)
            seg[((size_t)b * KK + m1) * NN + ncol] = acc1[v] * s;
    }
}

// ---------------------------------------------------------------------------
// K3: straight-through attn per pixel; denom[b,k] = sum_n attn
// grid: (N/256, B)
// ---------------------------------------------------------------------------
__global__ __launch_bounds__(256) void k3_attn_sums(
    const float* __restrict__ seg, float* __restrict__ denom)
{
    __shared__ float lsum[KK];
    const int b = blockIdx.y;
    const int n = blockIdx.x * 256 + threadIdx.x;
    const int tid = threadIdx.x, lane = tid & 31;
    if (tid < KK) lsum[tid] = 0.f;
    __syncthreads();

    float v[KK]; float mx = -INFINITY;
#pragma unroll
    for (int k = 0; k < KK; ++k) {
        v[k] = seg[((size_t)b * KK + k) * NN + n];
        mx = fmaxf(mx, v[k]);
    }
    int idx = 0; float bv = v[0];
#pragma unroll
    for (int k = 1; k < KK; ++k) if (v[k] > bv) { bv = v[k]; idx = k; }
    float se = 0.f;
#pragma unroll
    for (int k = 0; k < KK; ++k) { v[k] = __expf(v[k] - mx); se += v[k]; }
    const float inv = 1.f / se;
#pragma unroll
    for (int k = 0; k < KK; ++k) {
        float y = v[k] * inv;
        float h = (k == idx) ? 1.f : 0.f;
        float attn = (h - y) + y;          // straight-through forward value
        float r = attn;
        for (int off = 16; off > 0; off >>= 1) r += __shfl_down(r, off);
        if (lane == 0) atomicAdd(&lsum[k], r);
    }
    __syncthreads();
    if (tid < KK) atomicAdd(&denom[b * KK + tid], lsum[tid]);
}

// ---------------------------------------------------------------------------
// K4: out[b,n,c] = sum_k (attn/(denom+1)) * agg[b,k,c]   (bf16 WMMA)
// grid: (N/32, B); block handles 32 pixels x full C.
// ---------------------------------------------------------------------------
__global__ __launch_bounds__(256) void k4_out(
    const float* __restrict__ seg,
    const __bf16* __restrict__ aggBT,   // [B,C,KP] bf16
    const float* __restrict__ denom,
    float* __restrict__ out)            // [B,N,C]
{
    __shared__ __align__(16) __bf16 aggL[CC * KP];    // [c][k] 48 KB
    __shared__ __align__(16) __bf16 coeff[32 * KP];   // [n_local][k] 2 KB
    __shared__ float dsh[KK];
    const int b = blockIdx.y;
    const int n0 = blockIdx.x * 32;
    const int tid = threadIdx.x, lane = tid & 31, wave = tid >> 5;
    const int g = lane >> 4, lrow = lane & 15;

    if (tid < KK) dsh[tid] = denom[b * KK + tid] + 1.f;
    lds_copy_b128(aggL, aggBT + (size_t)b * CC * KP, CC * KP / 8, tid, 256);
    __syncthreads();

    if (tid < 32) {
        int n = n0 + tid;
        float v[KK]; float mx = -INFINITY;
#pragma unroll
        for (int k = 0; k < KK; ++k) {
            v[k] = seg[((size_t)b * KK + k) * NN + n];
            mx = fmaxf(mx, v[k]);
        }
        int idx = 0; float bv = v[0];
#pragma unroll
        for (int k = 1; k < KK; ++k) if (v[k] > bv) { bv = v[k]; idx = k; }
        float se = 0.f;
#pragma unroll
        for (int k = 0; k < KK; ++k) { v[k] = __expf(v[k] - mx); se += v[k]; }
        const float inv = 1.f / se;
#pragma unroll
        for (int k = 0; k < KK; ++k) {
            float y = v[k] * inv;
            float h = (k == idx) ? 1.f : 0.f;
            float attn = (h - y) + y;
            coeff[tid * KP + k] = (__bf16)(attn / dsh[k]);
        }
#pragma unroll
        for (int k = KK; k < KP; ++k) coeff[tid * KP + k] = (__bf16)0.f;
    }
    __syncthreads();

    v16bf a0 = load_frag16(&coeff[lrow * KP], g);
    v16bf a1 = load_frag16(&coeff[(16 + lrow) * KP], g);
    for (int ct = wave; ct < CC / 16; ct += 8) {
        v16bf bf = load_frag16(&aggL[(ct * 16 + lrow) * KP], g);
        v8f o0 = {}; v8f o1 = {};
        o0 = __builtin_amdgcn_wmma_f32_16x16x32_bf16(false, a0, false, bf,
                                                     (short)0, o0, false, false);
        o1 = __builtin_amdgcn_wmma_f32_16x16x32_bf16(false, a1, false, bf,
                                                     (short)0, o1, false, false);
        const int cbase = ct * 16 + lrow;
#pragma unroll
        for (int v = 0; v < 8; ++v) {
            int m = g * 8 + v;
            out[((size_t)b * NN + n0 + m) * CC + cbase] = o0[v];
            out[((size_t)b * NN + n0 + 16 + m) * CC + cbase] = o1[v];
        }
    }
}

// ---------------------------------------------------------------------------
extern "C" void kernel_launch(void* const* d_in, const int* in_sizes, int n_in,
                              void* d_out, int out_size, void* d_ws, size_t ws_size,
                              hipStream_t stream) {
    (void)in_sizes; (void)n_in; (void)out_size; (void)ws_size;
    const float* g_feat = (const float*)d_in[0];   // [B,1,C]
    const float* feat   = (const float*)d_in[1];   // [B,N,C]
    const float* tau    = (const float*)d_in[2];   // [1]
    const float* text   = (const float*)d_in[3];   // [T,C]
    float* out = (float*)d_out;                    // [B,N,C]

    float*  seg   = (float*)d_ws;                         // B*K*N fp32 (10.5 MB)
    __bf16* aggBF = (__bf16*)(seg + (size_t)BB * KK * NN);// B*KP*C bf16 (1.5 MB)
    __bf16* aggBT = aggBF + (size_t)BB * KP * CC;         // B*C*KP bf16 (1.5 MB)
    float*  denom = (float*)(aggBT + (size_t)BB * CC * KP); // B*K

    k1_topk_gather<<<BB, 256, 0, stream>>>(g_feat, text, aggBF, aggBT, denom);
    k2_seg_logit<<<dim3(NN / 128, BB), 256, 0, stream>>>(feat, aggBF, tau, seg);
    k3_attn_sums<<<dim3(NN / 256, BB), 256, 0, stream>>>(seg, denom);
    k4_out<<<dim3(NN / 32, BB), 256, 0, stream>>>(seg, aggBT, denom, out);
}
